// Metacontroller_85538568667439
// MI455X (gfx1250) — matrix operand order repaired and checked by reading, not compiled, not run on hardware
//
#include <hip/hip_runtime.h>

typedef __attribute__((ext_vector_type(16))) _Float16 v16h;
typedef __attribute__((ext_vector_type(8)))  _Float16 v8h;
typedef __attribute__((ext_vector_type(8)))  float    v8f;

#define BB 16
#define TT 256
#define DM 512
#define LATD 8
#define NHD 32
#define NSD 32
#define DSTATE 64
#define BT 4096          // BB*TT

// d_out layout (fp32, concatenated in return order)
#define OUT_MUS      0
#define OUT_LOGVARS  32768
#define OUT_BETAS    65536
#define OUT_ZS       69632
#define OUT_US       102400
#define OUT_KLS      67211264

__device__ __forceinline__ float sigmoidf_(float x){ return 1.f/(1.f+expf(-x)); }

__device__ __forceinline__ float wsum32(float v){
#pragma unroll
  for (int o=16;o>0;o>>=1) v += __shfl_xor(v, o, 32);
  return v;
}

// ---------------- WMMA fragment helpers (contiguous k = 16h+s order, wave32) -------
// A: 16x32 f16, row-major source (lda halves). lanes 0-15: M=lane, K 0-15
//                                              lanes16-31: M=lane-16, K 16-31
__device__ __forceinline__ v16h load_a_frag(const _Float16* A, int lda, int m0, int k0, int lane){
  int hh = lane >> 4;
  int m  = m0 + (lane & 15);
  const _Float16* p = A + (size_t)m*lda + k0 + hh*16;
  v8h lo = *(const v8h*)p;
  v8h hi = *(const v8h*)(p + 8);
  v16h r;
#pragma unroll
  for (int i=0;i<8;i++){ r[i]=lo[i]; r[i+8]=hi[i]; }
  return r;
}
// B: 32x16 f16 where B[k][n] = W[n][k], W row-major (N,K) (ldb halves).
// lane n = lane&15; lanes 0-15 hold K 0-15, lanes 16-31 hold K 16-31 -> one
// contiguous 32B load per lane.
__device__ __forceinline__ v16h load_b_frag(const _Float16* W, int ldb, int n0, int k0, int lane){
  int n = n0 + (lane & 15);
  int k = k0 + (lane >> 4)*16;
  return *(const v16h*)(W + (size_t)n*ldb + k);
}

// ---------------- prep kernels ---------------------------------------------
__global__ void k_cvt_f16(const float* __restrict__ src, _Float16* __restrict__ dst, int n){
  int i = blockIdx.x*blockDim.x + threadIdx.x;
  if (i < n) dst[i] = (_Float16)src[i];
}

// Wcat (224 x 512) f16: rows 0-95 gru_Wi, 96-127 sw_W1[:, :512],
// 128-191 enc_W1[:, :512], 192-223 in_W
__global__ void k_pack_wcat(const float* __restrict__ gruWi, const float* __restrict__ swW1,
                            const float* __restrict__ encW1, const float* __restrict__ inW,
                            _Float16* __restrict__ wcat){
  int i = blockIdx.x*blockDim.x + threadIdx.x;
  if (i >= 224*512) return;
  int row = i >> 9, k = i & 511;
  float v;
  if      (row <  96) v = gruWi[row*512 + k];
  else if (row < 128) v = swW1[(row-96)*552 + k];
  else if (row < 192) v = encW1[(row-128)*576 + k];
  else                v = inW[(row-192)*512 + k];
  wcat[i] = (_Float16)v;
}

// ---------------- fused input GEMM: Y(4096x224) = resid @ Wcat.T -----------
__global__ void k_gemm_in(const _Float16* __restrict__ A, const _Float16* __restrict__ W,
                          float* __restrict__ Y){
  int lane = threadIdx.x & 31;
  int gw   = blockIdx.x*(blockDim.x>>5) + (threadIdx.x>>5);   // 0..3583
  int mtile = gw / 14, ntile = gw % 14;
  int m0 = mtile*16, n0 = ntile*16;
  v8f c = {0.f,0.f,0.f,0.f,0.f,0.f,0.f,0.f};
  for (int k0=0; k0<512; k0+=32){
    v16h a = load_a_frag(A, 512, m0, k0, lane);
    v16h b = load_b_frag(W, 512, n0, k0, lane);
    c = __builtin_amdgcn_wmma_f32_16x16x32_f16(false, a, false, b, (short)0, c, false, false);
  }
  int col = n0 + (lane & 15);
  int mr  = m0 + ((lane >> 4) << 3);
#pragma unroll
  for (int r=0;r<8;r++) Y[(size_t)(mr + r)*224 + col] = c[r];
}

// ---------------- seq_embed pipeline (pass 0 = fwd, pass 1 = bwd) ----------
struct SeqP {
  const float *Dp, *Win, *Wout, *ln1b, *ln1g, *ln2b, *ln2g, *loglam, *m1W, *m1b, *m2W, *m2b;
};

__global__ void k_seq_init(const float* __restrict__ Y, const float* __restrict__ in_b,
                           float* __restrict__ xseq){
  int i = blockIdx.x*blockDim.x + threadIdx.x;   // 2*4096*32
  if (i >= 2*BT*NSD) return;
  int j = i & 31;
  int row = (i >> 5) & (BT-1);
  xseq[i] = Y[(size_t)row*224 + 192 + j] + in_b[j];
}

// ln1 + up = ln1(x) @ Win.T  (block=64, one row per block, rows are pass-major)
__global__ void k_seq_stage1(const float* __restrict__ xseq, float* __restrict__ lnx,
                             float* __restrict__ up, SeqP pf, SeqP pb){
  int row = blockIdx.x;              // 0..8191
  const SeqP p = (row >> 12) ? pb : pf;
  __shared__ float sln[NSD];
  int tid = threadIdx.x;
  if (tid < 32){
    float v = xseq[(size_t)row*32 + tid];
    float m = wsum32(v) * (1.f/32.f);
    float d = v - m;
    float var = wsum32(d*d) * (1.f/32.f);
    float ln = d * rsqrtf(var + 1e-5f) * p.ln1g[tid] + p.ln1b[tid];
    sln[tid] = ln;
    lnx[(size_t)row*32 + tid] = ln;
  }
  __syncthreads();
  float acc = 0.f;
#pragma unroll
  for (int i=0;i<32;i++) acc += sln[i] * p.Win[tid*32 + i];
  up[(size_t)row*64 + tid] = acc;
}

// LRU linear recurrence: one thread per (pass, b, state-dim); xs overwrites up
__global__ void k_seq_scan(float* __restrict__ up, SeqP pf, SeqP pb){
  int gid = blockIdx.x*blockDim.x + threadIdx.x;
  if (gid >= 2*BB*DSTATE) return;
  int d = gid & 63, b = (gid >> 6) & 15, pass = gid >> 10;
  const SeqP p = pass ? pb : pf;
  float lam = sigmoidf_(p.loglam[d]);
  float oml = 1.f - lam;
  float st = 0.f;
  float* col = up + ((size_t)(pass*BT + b*TT))*64 + d;
  if (pass == 0){
    for (int t=0;t<TT;t++){ st = lam*st + oml*col[(size_t)t*64]; col[(size_t)t*64] = st; }
  } else {
    for (int t=TT-1;t>=0;t--){ st = lam*st + oml*col[(size_t)t*64]; col[(size_t)t*64] = st; }
  }
}

// x += xs @ Wout.T + lnx * D
__global__ void k_seq_yres(float* __restrict__ xseq, const float* __restrict__ lnx,
                           const float* __restrict__ xs, SeqP pf, SeqP pb){
  int row = blockIdx.x;
  const SeqP p = (row >> 12) ? pb : pf;
  __shared__ float sxs[64];
  int tid = threadIdx.x;
  sxs[tid] = xs[(size_t)row*64 + tid];
  __syncthreads();
  if (tid < 32){
    float acc = 0.f;
#pragma unroll
    for (int k=0;k<64;k++) acc += sxs[k]*p.Wout[tid*64 + k];
    acc += lnx[(size_t)row*32 + tid] * p.Dp[tid];
    xseq[(size_t)row*32 + tid] += acc;
  }
}

// x += m2( relu( m1( ln2(x) ) ) )
__global__ void k_seq_mlp(float* __restrict__ xseq, SeqP pf, SeqP pb){
  int row = blockIdx.x;
  const SeqP p = (row >> 12) ? pb : pf;
  __shared__ float sln[32];
  __shared__ float shid[64];
  int tid = threadIdx.x;
  if (tid < 32){
    float v = xseq[(size_t)row*32 + tid];
    float m = wsum32(v) * (1.f/32.f);
    float d = v - m;
    float var = wsum32(d*d) * (1.f/32.f);
    sln[tid] = d * rsqrtf(var + 1e-5f) * p.ln2g[tid] + p.ln2b[tid];
  }
  __syncthreads();
  float acc = p.m1b[tid];
#pragma unroll
  for (int i=0;i<32;i++) acc += sln[i]*p.m1W[tid*32 + i];
  shid[tid] = fmaxf(acc, 0.f);
  __syncthreads();
  if (tid < 32){
    float o = p.m2b[tid];
#pragma unroll
    for (int k=0;k<64;k++) o += shid[k]*p.m2W[tid*64 + k];
    xseq[(size_t)row*32 + tid] += o;
  }
}

// s = [hf[:, -1], hb[:, 0]] @ out_W.T + out_b
__global__ void k_seq_out(const float* __restrict__ xseq, const float* __restrict__ outW,
                          const float* __restrict__ outb, float* __restrict__ sbuf){
  int b = blockIdx.x, j = threadIdx.x;
  float acc = outb[j];
#pragma unroll
  for (int k=0;k<64;k++){
    float c = (k < 32) ? xseq[(size_t)(b*TT + TT-1)*32 + k]
                       : xseq[(size_t)(BT + b*TT)*32 + (k-32)];
    acc += c*outW[j*64 + k];
  }
  sbuf[b*32 + j] = acc;
}

// ---------------- GRU scan: one wave per batch, Wh in registers ------------
__global__ void k_gru(const float* __restrict__ Y, const float* __restrict__ Wh,
                      const float* __restrict__ bh, const float* __restrict__ bi,
                      float* __restrict__ hseq){
  int b = blockIdx.x, j = threadIdx.x;   // j = 0..31
  float wr[32], wz[32], wn[32];
#pragma unroll
  for (int k=0;k<32;k++){ wr[k]=Wh[j*32+k]; wz[k]=Wh[(32+j)*32+k]; wn[k]=Wh[(64+j)*32+k]; }
  float bhr=bh[j], bhz=bh[32+j], bhn=bh[64+j];
  float bir=bi[j], biz=bi[32+j], bin_=bi[64+j];
  float h = 0.f;
  for (int t=0;t<TT;t++){
    size_t row = (size_t)b*TT + t;
    hseq[row*32 + j] = h;                  // pre-update h feeds enc/sw at step t
    float gir = Y[row*224 +      j] + bir;
    float giz = Y[row*224 + 32 + j] + biz;
    float gin = Y[row*224 + 64 + j] + bin_;
    float ar=bhr, az=bhz, an=bhn;
#pragma unroll
    for (int k=0;k<32;k++){
      float hk = __shfl(h, k, 32);
      ar += hk*wr[k]; az += hk*wz[k]; an += hk*wn[k];
    }
    float rg = sigmoidf_(gir + ar);
    float zg = sigmoidf_(giz + az);
    float n  = tanhf(gin + rg*an);
    h = (1.f - zg)*n + zg*h;
  }
}

// ---------------- encoder: mu/logvar per (b,t) row -------------------------
__global__ void k_enc(const float* __restrict__ Y, const float* __restrict__ hseq,
                      const float* __restrict__ sbuf,
                      const float* __restrict__ encW1, const float* __restrict__ encb1,
                      const float* __restrict__ encW2, const float* __restrict__ encb2,
                      float* __restrict__ out){
  int row = blockIdx.x, bidx = row >> 8;
  __shared__ float sh[32], ss[32], hid[64];
  int tid = threadIdx.x;
  if (tid < 32){ sh[tid] = hseq[(size_t)row*32 + tid]; ss[tid] = sbuf[bidx*32 + tid]; }
  __syncthreads();
  float acc = Y[(size_t)row*224 + 128 + tid] + encb1[tid];
  const float* wrow = encW1 + (size_t)tid*576;
#pragma unroll
  for (int k=0;k<32;k++) acc += sh[k]*wrow[512 + k];
#pragma unroll
  for (int k=0;k<32;k++) acc += ss[k]*wrow[544 + k];
  hid[tid] = fmaxf(acc, 0.f);
  __syncthreads();
  if (tid < 16){
    float o = encb2[tid];
#pragma unroll
    for (int u=0;u<64;u++) o += hid[u]*encW2[tid*64 + u];
    if (tid < 8) out[OUT_MUS     + (size_t)row*8 + tid]     = o;
    else         out[OUT_LOGVARS + (size_t)row*8 + (tid-8)] = o;
  }
}

// ---------------- z / beta / kl scan: one wave per batch -------------------
__global__ void k_zscan(const float* __restrict__ Y, const float* __restrict__ hseq,
                        const float* __restrict__ eps,
                        const float* __restrict__ swW1, const float* __restrict__ swb1,
                        const float* __restrict__ swW2, const float* __restrict__ swb2,
                        float* __restrict__ out){
  int b = blockIdx.x, j = threadIdx.x;   // j = 0..31
  float wh[32], wzp[8];
#pragma unroll
  for (int k=0;k<32;k++) wh[k]  = swW1[j*552 + 512 + k];
#pragma unroll
  for (int l=0;l<8;l++)  wzp[l] = swW1[j*552 + 544 + l];
  float b1 = swb1[j], w2 = swW2[j], b2 = swb2[0];
  float z = 0.f;                          // lanes >= 8 stay 0 forever
  for (int t=0;t<TT;t++){
    size_t row = (size_t)b*TT + t;
    float hj = hseq[row*32 + j];
    float a = Y[row*224 + 96 + j] + b1;
#pragma unroll
    for (int k=0;k<32;k++) a += __shfl(hj, k, 32)*wh[k];
#pragma unroll
    for (int l=0;l<8;l++)  a += __shfl(z,  l, 32)*wzp[l];
    float logit = wsum32(fmaxf(a, 0.f)*w2) + b2;
    float beta  = sigmoidf_(logit);
    if (j == 0) out[OUT_BETAS + row] = beta;
    float mu = out[OUT_MUS     + row*8 + (j & 7)];
    float lv = out[OUT_LOGVARS + row*8 + (j & 7)];
    float ep = eps[row*8 + (j & 7)];
    float mq = beta*mu + (1.f - beta)*z;
    float sq = beta*beta*expf(lv);
    float znew = mq + sqrtf(sq + 1e-8f)*ep;
    float diff = mq - z;
    float klt = logf(1.f+1e-8f) - logf(sq+1e-8f) + (sq + diff*diff)/(1.f+1e-8f) - 1.f;
    if (j >= 8) klt = 0.f;
    float kl = 0.5f * wsum32(klt);
    if (j < 8){ out[OUT_ZS + row*8 + j] = znew; z = znew; }
    if (j == 0) out[OUT_KLS + row] = kl;
  }
}

// ---------------- hyp stage 1: hid = relu(z @ W1.T + b1) -> f16 ------------
__global__ void k_hyp1(const float* __restrict__ out, const float* __restrict__ W1,
                       const float* __restrict__ b1, _Float16* __restrict__ hid_h){
  int i = blockIdx.x*blockDim.x + threadIdx.x;  // BT*32
  if (i >= BT*32) return;
  int row = i >> 5, j = i & 31;
  float acc = b1[j];
#pragma unroll
  for (int l=0;l<8;l++) acc += out[OUT_ZS + (size_t)row*8 + l]*W1[j*8 + l];
  hid_h[i] = (_Float16)fmaxf(acc, 0.f);
}

// ---------------- hyp GEMM: (4096x32) @ (32x16384) -> 256 MB of us ---------
__global__ void k_hyp_gemm(const _Float16* __restrict__ A, const _Float16* __restrict__ W,
                           const float* __restrict__ b2, float* __restrict__ us){
  int lane = threadIdx.x & 31;
  int gw   = blockIdx.x*(blockDim.x>>5) + (threadIdx.x>>5);  // 0..32767
  int mtile = gw >> 7;
  int ng    = gw & 127;
  int m0 = mtile*16;
  v16h a = load_a_frag(A, 32, m0, 0, lane);
  int col = lane & 15;
  int mr  = m0 + ((lane >> 4) << 3);
  for (int nt=0; nt<8; nt++){
    int n0 = (ng*8 + nt)*16;
    v16h bf = load_b_frag(W, 32, n0, 0, lane);
    v8f c = {0.f,0.f,0.f,0.f,0.f,0.f,0.f,0.f};
    c = __builtin_amdgcn_wmma_f32_16x16x32_f16(false, a, false, bf, (short)0, c, false, false);
    float bias = b2[n0 + col];
#pragma unroll
    for (int r=0;r<8;r++)
      us[(size_t)(mr + r)*16384 + n0 + col] = c[r] + bias;
  }
}

// ---------------- host side ------------------------------------------------
extern "C" void kernel_launch(void* const* d_in, const int* in_sizes, int n_in,
                              void* d_out, int out_size, void* d_ws, size_t ws_size,
                              hipStream_t stream) {
  auto F = [&](int i){ return (const float*)d_in[i]; };
  // pytree (sorted-key) input order
  const float* resid  = F(0);
  const float* eps    = F(1);
  const float* encW1  = F(2);
  const float* encW2  = F(3);
  const float* encb1  = F(4);
  const float* encb2  = F(5);
  const float* gruWh  = F(6);
  const float* gruWi  = F(7);
  const float* grubh  = F(8);
  const float* grubi  = F(9);
  const float* hypW1  = F(10);
  const float* hypW2  = F(11);
  const float* hypb1  = F(12);
  const float* hypb2  = F(13);
  auto mkSeq = [&](int base){
    SeqP p;
    p.Dp = F(base+0); p.Win = F(base+1); p.Wout = F(base+2);
    p.ln1b = F(base+3); p.ln1g = F(base+4); p.ln2b = F(base+5); p.ln2g = F(base+6);
    p.loglam = F(base+7); p.m1W = F(base+8); p.m1b = F(base+9); p.m2W = F(base+10); p.m2b = F(base+11);
    return p;
  };
  SeqP pb = mkSeq(14);     // seq.bwd
  SeqP pf = mkSeq(26);     // seq.fwd
  const float* in_b  = F(39);
  const float* outW  = F(40);
  const float* outb  = F(41);
  const float* swW1  = F(42);
  const float* swW2  = F(43);
  const float* swb1  = F(44);
  const float* swb2  = F(45);

  char* ws = (char*)d_ws;
  _Float16* resid_h = (_Float16*)(ws + 0);           // 4,194,304 B
  _Float16* wcat_h  = (_Float16*)(ws + 4194304);     //   229,376 B
  float*    Y       = (float*)   (ws + 4423680);     // 3,670,016 B
  float*    sbuf    = (float*)   (ws + 8093696);     //     2,048 B
  float*    hseq    = (float*)   (ws + 8095744);     //   524,288 B
  _Float16* hid_h   = (_Float16*)(ws + 8620032);     //   262,144 B
  _Float16* W2_h    = (_Float16*)(ws + 8882176);     // 1,048,576 B
  float*    xseq    = (float*)   (ws + 9930752);     // 1,048,576 B
  float*    lnx     = (float*)   (ws + 10979328);    // 1,048,576 B
  float*    up      = (float*)   (ws + 12027904);    // 2,097,152 B (xs overwrites)

  float* out = (float*)d_out;

  k_cvt_f16  <<<8192, 256, 0, stream>>>(resid, resid_h, BT*DM);
  k_cvt_f16  <<<2048, 256, 0, stream>>>(hypW2, W2_h, 16384*32);
  k_pack_wcat<<<448,  256, 0, stream>>>(gruWi, swW1, encW1, F(38) /*in_W*/, wcat_h);

  k_gemm_in  <<<448,  256, 0, stream>>>(resid_h, wcat_h, Y);

  k_seq_init  <<<1024, 256, 0, stream>>>(Y, in_b, xseq);
  k_seq_stage1<<<8192, 64,  0, stream>>>(xseq, lnx, up, pf, pb);
  k_seq_scan  <<<8,    256, 0, stream>>>(up, pf, pb);
  k_seq_yres  <<<8192, 64,  0, stream>>>(xseq, lnx, up, pf, pb);
  k_seq_mlp   <<<8192, 64,  0, stream>>>(xseq, pf, pb);
  k_seq_out   <<<16,   32,  0, stream>>>(xseq, outW, outb, sbuf);

  k_gru   <<<16,   32, 0, stream>>>(Y, gruWh, grubh, grubi, hseq);
  k_enc   <<<4096, 64, 0, stream>>>(Y, hseq, sbuf, encW1, encb1, encW2, encb2, out);
  k_zscan <<<16,   32, 0, stream>>>(Y, hseq, eps, swW1, swb1, swW2, swb2, out);

  k_hyp1    <<<512,  256, 0, stream>>>(out, hypW1, hypb1, hid_h);
  k_hyp_gemm<<<4096, 256, 0, stream>>>(hid_h, W2_h, hypb2, out + OUT_US);
}
